// GCNLayer_24524263260207
// MI455X (gfx1250) — matrix operand order repaired and checked by reading, not compiled, not run on hardware
//
#include <hip/hip_runtime.h>
#include <cmath>

typedef __attribute__((ext_vector_type(16))) _Float16 v16h;
typedef __attribute__((ext_vector_type(8)))  _Float16 v8h;
typedef __attribute__((ext_vector_type(8)))  float    v8f;

#define F_DIM 128
#define D_DIM 64
#define STATS_BLOCKS 256
#define BN_EPS 1e-3f
#define LDA_H 136   // half-stride per A row in LDS (128 + 8 pad, multiple of 8 for b128 align)

// ---------------------------------------------------------------------------
// Stage 1: per-block partial column sums / sums-of-squares (deterministic,
// no float atomics). thread = column -> coalesced 512B row reads.
// ---------------------------------------------------------------------------
__global__ void __launch_bounds__(F_DIM)
k_colstats_partial(const float* __restrict__ x, float* __restrict__ psum,
                   float* __restrict__ psumsq, int N) {
    int f = threadIdx.x;
    int b = blockIdx.x;
    int rows_per = (N + gridDim.x - 1) / gridDim.x;
    int r0 = b * rows_per;
    int r1 = r0 + rows_per; if (r1 > N) r1 = N;
    float s = 0.f, s2 = 0.f;
    for (int r = r0; r < r1; ++r) {
        float v = x[(size_t)r * F_DIM + f];
        s += v; s2 += v * v;
    }
    psum[b * F_DIM + f]   = s;
    psumsq[b * F_DIM + f] = s2;
}

// Stage 2: finalize mean / rstd per column.
__global__ void __launch_bounds__(F_DIM)
k_colstats_final(const float* __restrict__ psum, const float* __restrict__ psumsq,
                 float* __restrict__ meanv, float* __restrict__ rstdv, int N) {
    int f = threadIdx.x;
    float s = 0.f, s2 = 0.f;
    for (int b = 0; b < STATS_BLOCKS; ++b) {
        s  += psum[b * F_DIM + f];
        s2 += psumsq[b * F_DIM + f];
    }
    float m   = s / (float)N;
    float var = s2 / (float)N - m * m;
    meanv[f] = m;
    rstdv[f] = rsqrtf(var + BN_EPS);
}

// Fold BN into weights: W'[f][d] = rstd[f]*W[f][d]; bias[d] = -sum_f mean[f]*W'[f][d]
// => h = x @ W' + bias  (x read raw, no normalized copy)
__global__ void __launch_bounds__(D_DIM)
k_fold(const float* __restrict__ kernelW, const float* __restrict__ meanv,
       const float* __restrict__ rstdv, float* __restrict__ wfold,
       float* __restrict__ bias) {
    int d = threadIdx.x;
    float acc = 0.f;
    for (int f = 0; f < F_DIM; ++f) {
        float wf = rstdv[f] * kernelW[f * D_DIM + d];
        wfold[f * D_DIM + d] = wf;
        acc += meanv[f] * wf;
    }
    bias[d] = -acc;
}

// ---------------------------------------------------------------------------
// Pre-pack B into the exact per-lane WMMA fragment layout as f16.
// bpack[((wv*4 + s)*32 + lane)*16 + i], i = 0..15 halves, contiguous 32B/lane.
// B layout used here (contiguous K, matching A): lane = column n; lanes 0-15 hold K=0..15, 16-31 hold K=16..31.
// 512 threads: one per (colTile wv, kStep s, lane). Total 16KB, L2-resident.
// ---------------------------------------------------------------------------
__global__ void __launch_bounds__(512)
k_packB(const float* __restrict__ wfold, _Float16* __restrict__ bpack) {
    int t    = threadIdx.x;        // 0..511
    int wv   = t >> 7;             // column tile 0..3
    int s    = (t >> 5) & 3;       // K step 0..3
    int lane = t & 31;
    int m    = lane & 15;
    int hi   = lane >> 4;
    int n    = wv * 16 + m;
    _Float16* dst = bpack + (size_t)((wv * 4 + s) * 32 + lane) * 16;
    #pragma unroll
    for (int v = 0; v < 8; ++v) {
        int k = s * 32 + hi * 16 + 2 * v;
        dst[2 * v]     = (_Float16)wfold[k * D_DIM + n];
        dst[2 * v + 1] = (_Float16)wfold[(k + 1) * D_DIM + n];
    }
}

// ---------------------------------------------------------------------------
// WMMA GEMM: h[N,64] = x[N,128] @ W' + bias
// Block = 128 threads = 4 waves; block tile = 32 rows x 64 cols.
// Wave w owns column tile n0=16w and both 16-row halves (c0: rows 0-15,
// c1: rows 16-31) so each B fragment feeds two WMMAs.
// Bias is splat-folded into the accumulator init (all 8 C elements of a lane
// share column n). Full tiles take a branchless store path.
// ---------------------------------------------------------------------------
__global__ void __launch_bounds__(128)
k_gemm_wmma(const float* __restrict__ x, const _Float16* __restrict__ bpack,
            const float* __restrict__ bias, float* __restrict__ h, int N) {
    __shared__ __align__(16) _Float16 lAh[32 * LDA_H];
    const int row0 = blockIdx.x * 32;
    const int t = threadIdx.x;

    // cooperative stage of A: 32 rows x 128 floats = 1024 float4, 8 per thread,
    // converting f32 -> f16 once here (v_cvt_pk_f16_f32 out of the hot loop)
    #pragma unroll
    for (int i = 0; i < 8; ++i) {
        int idx = t + 128 * i;          // float4 slot 0..1023 (32 per row)
        int r   = idx >> 5;
        int c   = (idx & 31) << 2;
        int row = row0 + r; if (row >= N) row = N - 1;   // clamp (N%32==0 normally)
        const float4 v = *(const float4*)(x + (size_t)row * F_DIM + c);
        _Float16* dst = &lAh[r * LDA_H + c];
        dst[0] = (_Float16)v.x; dst[1] = (_Float16)v.y;
        dst[2] = (_Float16)v.z; dst[3] = (_Float16)v.w;
    }
    __syncthreads();

    const int lane = t & 31;
    const int wv   = t >> 5;       // wave id -> column tile
    const int m    = lane & 15;
    const int hi   = lane >> 4;    // lane half (0/1)
    const int n    = wv * 16 + m;

    const float bv = bias[n];
    v8f c0 = {bv, bv, bv, bv, bv, bv, bv, bv};
    v8f c1 = c0;

    #pragma unroll
    for (int s = 0; s < 4; ++s) {
        // B fragment: one contiguous 32B read per lane (L2-resident)
        const v16h B = *(const v16h*)(bpack + (size_t)((wv * 4 + s) * 32 + lane) * 16);

        // A fragment, contiguous K order matching bpack: lane half picks kbase (0 or 16);
        // halves [0..15] = K kbase..kbase+15.
        const int ka = s * 32 + hi * 16;
        v8h a0 = *(const v8h*)&lAh[m * LDA_H + ka];
        v8h a1 = *(const v8h*)&lAh[m * LDA_H + ka + 8];
        v16h A0 = __builtin_shufflevector(a0, a1, 0,1,2,3,4,5,6,7,8,9,10,11,12,13,14,15);
        v8h b0 = *(const v8h*)&lAh[(m + 16) * LDA_H + ka];
        v8h b1 = *(const v8h*)&lAh[(m + 16) * LDA_H + ka + 8];
        v16h A1 = __builtin_shufflevector(b0, b1, 0,1,2,3,4,5,6,7,8,9,10,11,12,13,14,15);

        c0 = __builtin_amdgcn_wmma_f32_16x16x32_f16(false, A0, false, B, (short)0, c0, false, false);
        c1 = __builtin_amdgcn_wmma_f32_16x16x32_f16(false, A1, false, B, (short)0, c1, false, false);
    }

    // C/D layout: VGPR r -> row (r + 8*hi), col n. Consecutive r = consecutive rows
    // -> stores become immediate-offset global_store_b32 off one base per lane.
    if (row0 + 32 <= N) {   // block-uniform fast path (always true when N%32==0)
        float* hp0 = h + (size_t)(row0 + 8 * hi) * D_DIM + n;
        float* hp1 = h + (size_t)(row0 + 16 + 8 * hi) * D_DIM + n;
        #pragma unroll
        for (int r = 0; r < 8; ++r) {
            hp0[(size_t)r * D_DIM] = c0[r];
            hp1[(size_t)r * D_DIM] = c1[r];
        }
    } else {
        #pragma unroll
        for (int r = 0; r < 8; ++r) {
            int rowA = row0 + r + 8 * hi;
            if (rowA < N) h[(size_t)rowA * D_DIM + n] = c0[r];
            int rowB = row0 + 16 + r + 8 * hi;
            if (rowB < N) h[(size_t)rowB * D_DIM + n] = c1[r];
        }
    }
}

// ---------------------------------------------------------------------------
// Edge softmax (segment max / exp / segment sum)
// ---------------------------------------------------------------------------
__global__ void k_rowinit(unsigned* __restrict__ rowmax_u, float* __restrict__ rowsum, int N) {
    int i = blockIdx.x * blockDim.x + threadIdx.x;
    if (i < N) { rowmax_u[i] = 0u; rowsum[i] = 0.f; }
}

// edge_vals in [0,1) => non-negative floats => uint bit pattern is order-preserving
__global__ void k_edge_max(const float* __restrict__ ev, const int* __restrict__ rows,
                           unsigned* __restrict__ rowmax_u, int E) {
    int i = blockIdx.x * blockDim.x + threadIdx.x;
    if (i < E) atomicMax(&rowmax_u[rows[i]], __float_as_uint(ev[i]));
}

__global__ void k_edge_expsum(const float* __restrict__ ev, const int* __restrict__ rows,
                              const float* __restrict__ rowmax, float* __restrict__ eexp,
                              float* __restrict__ rowsum, int E) {
    int i = blockIdx.x * blockDim.x + threadIdx.x;
    if (i < E) {
        int r = rows[i];
        float e = expf(ev[i] - rowmax[r]);
        eexp[i] = e;
        atomicAdd(&rowsum[r], e);
    }
}

__global__ void k_zero(float* __restrict__ p, int n) {
    int i = blockIdx.x * blockDim.x + threadIdx.x;
    if (i < n) p[i] = 0.f;
}

// ---------------------------------------------------------------------------
// Scatter: one wave32 per edge, 2 floats per lane.
// h (25.6MB) and out (25.6MB) both fit in the 192MB L2 -> gathers/atomics on-chip.
// ---------------------------------------------------------------------------
__global__ void __launch_bounds__(256)
k_scatter(const float* __restrict__ eexp, const float* __restrict__ rowsum,
          const int* __restrict__ rows, const int* __restrict__ cols,
          const float* __restrict__ h, float* __restrict__ out, int E) {
    int wave = threadIdx.x >> 5;
    int lane = threadIdx.x & 31;
    int edge = blockIdx.x * 8 + wave;
    if (edge >= E) return;
    int r   = rows[edge];
    int col = cols[edge];
    float a = eexp[edge] / rowsum[r];
    const float2 hv = *(const float2*)(h + (size_t)col * D_DIM + 2 * lane);
    atomicAdd(out + (size_t)r * D_DIM + 2 * lane,     a * hv.x);
    atomicAdd(out + (size_t)r * D_DIM + 2 * lane + 1, a * hv.y);
}

__global__ void k_tanh(float* __restrict__ out, int n) {
    int i = blockIdx.x * blockDim.x + threadIdx.x;
    if (i < n) out[i] = tanhf(out[i]);
}

// ---------------------------------------------------------------------------
extern "C" void kernel_launch(void* const* d_in, const int* in_sizes, int n_in,
                              void* d_out, int out_size, void* d_ws, size_t ws_size,
                              hipStream_t stream) {
    const float* x       = (const float*)d_in[0];
    const float* kernelW = (const float*)d_in[1];
    const float* ev      = (const float*)d_in[2];
    const int*   rows    = (const int*)d_in[3];
    const int*   cols    = (const int*)d_in[4];

    const int N = in_sizes[0] / F_DIM;
    const int E = in_sizes[2];
    float* out = (float*)d_out;

    // workspace layout (floats); bpack offset is 32B-aligned by construction
    float* ws     = (float*)d_ws;
    float* psum   = ws;                                   // STATS_BLOCKS*128
    float* psumsq = psum   + STATS_BLOCKS * F_DIM;        // STATS_BLOCKS*128
    float* meanv  = psumsq + STATS_BLOCKS * F_DIM;        // 128
    float* rstdv  = meanv  + F_DIM;                       // 128
    float* wfold  = rstdv  + F_DIM;                       // 128*64
    float* bias   = wfold  + F_DIM * D_DIM;               // 64
    float* bpackf = bias   + D_DIM;                       // 4096 floats = 8192 halves
    _Float16* bpack = (_Float16*)bpackf;
    float* h      = bpackf + (F_DIM * D_DIM) / 2;         // N*64
    float* rowmax = h      + (size_t)N * D_DIM;           // N
    float* rowsum = rowmax + N;                           // N
    float* eexp   = rowsum + N;                           // E

    // 1) BatchNorm stats (two-stage deterministic reduction) + fold into weights
    k_colstats_partial<<<STATS_BLOCKS, F_DIM, 0, stream>>>(x, psum, psumsq, N);
    k_colstats_final<<<1, F_DIM, 0, stream>>>(psum, psumsq, meanv, rstdv, N);
    k_fold<<<1, D_DIM, 0, stream>>>(kernelW, meanv, rstdv, wfold, bias);
    k_packB<<<1, 512, 0, stream>>>(wfold, bpack);

    // 2) WMMA GEMM: h = x @ W' + bias  (32-row x 64-col block tiles)
    k_gemm_wmma<<<(N + 31) / 32, 128, 0, stream>>>(x, bpack, bias, h, N);

    // 3) edge softmax
    k_rowinit<<<(N + 255) / 256, 256, 0, stream>>>((unsigned*)rowmax, rowsum, N);
    k_edge_max<<<(E + 255) / 256, 256, 0, stream>>>(ev, rows, (unsigned*)rowmax, E);
    k_edge_expsum<<<(E + 255) / 256, 256, 0, stream>>>(ev, rows, rowmax, eexp, rowsum, E);

    // 4) scatter-add of attention-weighted messages, then tanh
    k_zero<<<((size_t)N * D_DIM + 255) / 256, 256, 0, stream>>>(out, N * D_DIM);
    k_scatter<<<(E + 7) / 8, 256, 0, stream>>>(eexp, rowsum, rows, cols, h, out, E);
    k_tanh<<<((size_t)N * D_DIM + 255) / 256, 256, 0, stream>>>(out, N * D_DIM);
}
